// MSDeformAttnTransformerEncoderOnly_72146860638800
// MI455X (gfx1250) — matrix operand order repaired and checked by reading, not compiled, not run on hardware
//
#include <hip/hip_runtime.h>

typedef _Float16 h16;
typedef _Float16 v8h  __attribute__((ext_vector_type(8)));
typedef _Float16 v16h __attribute__((ext_vector_type(16)));
typedef float    v8f  __attribute__((ext_vector_type(8)));

static constexpr int BS = 2, DM = 256, DFF = 1024, NH = 8, NLV = 4, NPT = 4, NLAYERS = 6, HD = 32;
static constexpr int NCLIP = 41, NBUS = 32;
static constexpr int HWALL = 12240;
static constexpr int LTOT  = HWALL + NCLIP + NBUS;   // 12313
static constexpr int RTOT  = BS * LTOT;              // 24626
static constexpr int RPAD  = ((RTOT + 127) / 128) * 128;  // 24704

static constexpr int c_LH[4] = {96, 48, 24, 12};
static constexpr int c_LW[4] = {96, 48, 24, 12};
static constexpr int c_LS[4] = {0, 9216, 11520, 12096};

// ---------------------------------------------------------------------------
// Weight convert: dst[n*K + k] = (f16) src[k*N + n]   (transpose to [N][K])
// ---------------------------------------------------------------------------
__global__ void k_wcvt(const float* __restrict__ src, h16* __restrict__ dst, int K, int N) {
  int i = blockIdx.x * blockDim.x + threadIdx.x;
  if (i >= K * N) return;
  int n = i / K, k = i - n * K;
  dst[i] = (h16)src[(size_t)k * N + n];
}

// ---------------------------------------------------------------------------
// Build pos stream: spatial -> pos+level_embed, bus -> memory_pos, clip -> 0
// ---------------------------------------------------------------------------
__global__ void k_build_pos(const float* __restrict__ p0, const float* __restrict__ p1,
                            const float* __restrict__ p2, const float* __restrict__ p3,
                            const float* __restrict__ lvl_emb, const float* __restrict__ mpos,
                            float* __restrict__ POSB) {
  size_t i = (size_t)blockIdx.x * blockDim.x + threadIdx.x;
  if (i >= (size_t)RTOT * DM) return;
  int c = (int)(i & (DM - 1));
  int r = (int)(i >> 8);
  int b = r / LTOT, t = r - b * LTOT;
  float v = 0.f;
  if (t < HWALL) {
    int lv = (t < 9216) ? 0 : (t < 11520) ? 1 : (t < 12096) ? 2 : 3;
    int local = t - c_LS[lv];
    int hw = c_LH[lv] * c_LW[lv];
    const float* p = (lv == 0) ? p0 : (lv == 1) ? p1 : (lv == 2) ? p2 : p3;
    v = p[((size_t)b * DM + c) * hw + local] + lvl_emb[lv * DM + c];
  } else if (t >= HWALL + NCLIP) {
    v = mpos[(t - (HWALL + NCLIP)) * DM + c];
  }
  POSB[i] = v;
}

// ---------------------------------------------------------------------------
// Build initial token stream X: [srcs | obj_text@clip_W+clip_b | memory_bus]
// ---------------------------------------------------------------------------
__global__ void k_build_x0(const float* __restrict__ s0, const float* __restrict__ s1,
                           const float* __restrict__ s2, const float* __restrict__ s3,
                           const float* __restrict__ mbus, const float* __restrict__ obj_text,
                           const float* __restrict__ clip_W, const float* __restrict__ clip_b,
                           float* __restrict__ X) {
  size_t i = (size_t)blockIdx.x * blockDim.x + threadIdx.x;
  if (i >= (size_t)RTOT * DM) return;
  int c = (int)(i & (DM - 1));
  int r = (int)(i >> 8);
  int b = r / LTOT, t = r - b * LTOT;
  float v;
  if (t < HWALL) {
    int lv = (t < 9216) ? 0 : (t < 11520) ? 1 : (t < 12096) ? 2 : 3;
    int local = t - c_LS[lv];
    int hw = c_LH[lv] * c_LW[lv];
    const float* p = (lv == 0) ? s0 : (lv == 1) ? s1 : (lv == 2) ? s2 : s3;
    v = p[((size_t)b * DM + c) * hw + local];
  } else if (t < HWALL + NCLIP) {
    int ti = t - HWALL;
    float acc = clip_b[c];
    const float* ot = obj_text + (size_t)ti * 512;
    for (int k = 0; k < 512; k++) acc += ot[k] * clip_W[(size_t)k * DM + c];
    v = acc;
  } else {
    v = mbus[(t - (HWALL + NCLIP)) * DM + c];
  }
  X[i] = v;
}

// ---------------------------------------------------------------------------
// Per-layer prep: Xh = f16(X), Qh = f16(X + pos)
// ---------------------------------------------------------------------------
__global__ void k_prep(const float* __restrict__ X, const float* __restrict__ POSB,
                       h16* __restrict__ Xh, h16* __restrict__ Qh) {
  size_t i = (size_t)blockIdx.x * blockDim.x + threadIdx.x;
  if (i >= (size_t)RTOT * DM) return;
  float x = X[i];
  Xh[i] = (h16)x;
  Qh[i] = (h16)(x + POSB[i]);
}

// ---------------------------------------------------------------------------
// WMMA GEMM:  C[R x N] = A[R x K] (f16 row-major) * Wt[N x K] (f16, W^T) + bias
// Each wave owns a 32x64 output tile: 2 A-fragments x 4 B-fragments -> 8 WMMA
// per 12 b128 loads per lane (register blocking for FLOP/byte).
// A fragment (16x32 f16):  lanes 0-15 row M=lane, K kk..kk+15
//                          lanes 16-31 row M=lane-16, K kk+16..kk+31
// B fragment (32x16 f16):  lanes 0-15 col N=lane, K kk..kk+15
//                          lanes 16-31 col N=lane-16, K kk+16..kk+31
// Waves in a block tile consecutive 64-col groups of the same 32-row strip.
// EPI: 0 = write f32, 1 = ReLU + write f16
// ---------------------------------------------------------------------------
template <int K, int N, int EPI>
__global__ __launch_bounds__(128) void k_gemm(const h16* __restrict__ A,
                                              const h16* __restrict__ Wt,
                                              const float* __restrict__ bias,
                                              float* __restrict__ Cf, h16* __restrict__ Ch,
                                              int rows) {
  const int lane = threadIdx.x & 31;
  const int wv   = threadIdx.x >> 5;
  const int hi   = lane >> 4;   // 0 | 1 (half-wave)
  const int l15  = lane & 15;
  const int wavesPerBlock = (int)(blockDim.x >> 5);
  const int colBase = (blockIdx.y * wavesPerBlock + wv) * 64;
  const int rowBase = blockIdx.x * 32;

  v8f acc[2][4];
  {
    const float b0 = bias[colBase + l15];
    const float b1 = bias[colBase + 16 + l15];
    const float b2 = bias[colBase + 32 + l15];
    const float b3 = bias[colBase + 48 + l15];
#pragma unroll
    for (int m = 0; m < 2; m++) {
      acc[m][0] = (v8f){b0, b0, b0, b0, b0, b0, b0, b0};
      acc[m][1] = (v8f){b1, b1, b1, b1, b1, b1, b1, b1};
      acc[m][2] = (v8f){b2, b2, b2, b2, b2, b2, b2, b2};
      acc[m][3] = (v8f){b3, b3, b3, b3, b3, b3, b3, b3};
    }
  }

  const h16* ap0 = A + (size_t)(rowBase + l15) * K + hi * 16;
  const h16* ap1 = ap0 + (size_t)16 * K;
  const h16* bp0 = Wt + (size_t)(colBase + l15) * K + hi * 16;
  const h16* bp1 = bp0 + (size_t)16 * K;
  const h16* bp2 = bp0 + (size_t)32 * K;
  const h16* bp3 = bp0 + (size_t)48 * K;

#pragma unroll 4
  for (int kk = 0; kk < K; kk += 32) {
    union { v16h v; v8h h[2]; } af[2], bf[4];
    af[0].h[0] = *(const v8h*)(ap0 + kk);
    af[0].h[1] = *(const v8h*)(ap0 + kk + 8);
    af[1].h[0] = *(const v8h*)(ap1 + kk);
    af[1].h[1] = *(const v8h*)(ap1 + kk + 8);
    bf[0].h[0] = *(const v8h*)(bp0 + kk);
    bf[0].h[1] = *(const v8h*)(bp0 + kk + 8);
    bf[1].h[0] = *(const v8h*)(bp1 + kk);
    bf[1].h[1] = *(const v8h*)(bp1 + kk + 8);
    bf[2].h[0] = *(const v8h*)(bp2 + kk);
    bf[2].h[1] = *(const v8h*)(bp2 + kk + 8);
    bf[3].h[0] = *(const v8h*)(bp3 + kk);
    bf[3].h[1] = *(const v8h*)(bp3 + kk + 8);
#pragma unroll
    for (int n = 0; n < 4; n++) {
      acc[0][n] = __builtin_amdgcn_wmma_f32_16x16x32_f16(false, af[0].v, false, bf[n].v,
                                                         (short)0, acc[0][n], false, false);
      acc[1][n] = __builtin_amdgcn_wmma_f32_16x16x32_f16(false, af[1].v, false, bf[n].v,
                                                         (short)0, acc[1][n], false, false);
    }
  }

#pragma unroll
  for (int m = 0; m < 2; m++) {
    const int rb = rowBase + m * 16 + hi * 8;
#pragma unroll
    for (int n = 0; n < 4; n++) {
      const int col = colBase + n * 16 + l15;
#pragma unroll
      for (int j = 0; j < 8; j++) {
        int r = rb + j;
        if (r < rows) {
          if (EPI == 0) {
            Cf[(size_t)r * N + col] = acc[m][n][j];
          } else {
            float v = acc[m][n][j] > 0.f ? acc[m][n][j] : 0.f;
            Ch[(size_t)r * N + col] = (h16)v;
          }
        }
      }
    }
  }
}

// ---------------------------------------------------------------------------
// Softmax over the 16 (level,point) logits per (row, head), in place
// ---------------------------------------------------------------------------
__global__ void k_softmax(float* __restrict__ ATT) {
  int i = blockIdx.x * blockDim.x + threadIdx.x;
  if (i >= RTOT * NH) return;
  int r = i >> 3, h = i & 7;
  float* p = ATT + (size_t)r * 128 + h * 16;
  float m = -1e30f;
  for (int j = 0; j < 16; j++) m = fmaxf(m, p[j]);
  float e[16], s = 0.f;
  for (int j = 0; j < 16; j++) { e[j] = __expf(p[j] - m); s += e[j]; }
  float inv = 1.f / s;
  for (int j = 0; j < 16; j++) p[j] = e[j] * inv;
}

// ---------------------------------------------------------------------------
// Deformable sampling: block = token row, wave = head, lane = head channel
// ---------------------------------------------------------------------------
__global__ __launch_bounds__(256) void k_sample(const float* __restrict__ VAL,
                                                const float* __restrict__ OFFB,
                                                const float* __restrict__ ATT,
                                                h16* __restrict__ AH) {
  int r = blockIdx.x;
  int h = threadIdx.x >> 5;
  int d = threadIdx.x & 31;
  int b = r / LTOT, t = r - b * LTOT;

  float rx = 0.f, ry = 0.f;
  if (t < HWALL) {
    int lv = (t < 9216) ? 0 : (t < 11520) ? 1 : (t < 12096) ? 2 : 3;
    int local = t - c_LS[lv];
    int W = c_LW[lv], H = c_LH[lv];
    int yy = local / W, xx = local - yy * W;
    rx = (xx + 0.5f) / W;
    ry = (yy + 0.5f) / H;
  }

  const float* offp = OFFB + (size_t)r * DM + h * (NLV * NPT * 2);
  const float* attp = ATT + (size_t)r * 128 + h * 16;
  float acc = 0.f;

#pragma unroll
  for (int l = 0; l < NLV; l++) {
    const int W = c_LW[l], H = c_LH[l];
    const float* vbase = VAL + ((size_t)(b * LTOT + c_LS[l])) * DM + h * HD + d;
#pragma unroll
    for (int p = 0; p < NPT; p++) {
      float ox = offp[(l * NPT + p) * 2 + 0];
      float oy = offp[(l * NPT + p) * 2 + 1];
      float x = (rx + ox / W) * W - 0.5f;
      float y = (ry + oy / H) * H - 0.5f;
      float x0 = floorf(x), y0 = floorf(y);
      float dx = x - x0, dy = y - y0;
      int ix = (int)x0, iy = (int)y0;
      float s = 0.f;
      { int cx = ix,     cy = iy;     float w = (1.f - dx) * (1.f - dy);
        if (cx >= 0 && cx < W && cy >= 0 && cy < H) s += w * vbase[(size_t)(cy * W + cx) * DM]; }
      { int cx = ix + 1, cy = iy;     float w = dx * (1.f - dy);
        if (cx >= 0 && cx < W && cy >= 0 && cy < H) s += w * vbase[(size_t)(cy * W + cx) * DM]; }
      { int cx = ix,     cy = iy + 1; float w = (1.f - dx) * dy;
        if (cx >= 0 && cx < W && cy >= 0 && cy < H) s += w * vbase[(size_t)(cy * W + cx) * DM]; }
      { int cx = ix + 1, cy = iy + 1; float w = dx * dy;
        if (cx >= 0 && cx < W && cy >= 0 && cy < H) s += w * vbase[(size_t)(cy * W + cx) * DM]; }
      acc += attp[l * NPT + p] * s;
    }
  }
  AH[(size_t)r * DM + h * HD + d] = (h16)acc;
}

// ---------------------------------------------------------------------------
// Residual add + LayerNorm (block per row, LDS tree reduction).
// ---------------------------------------------------------------------------
__global__ __launch_bounds__(256) void k_layernorm(float* __restrict__ X,
                                                   const float* __restrict__ resid,
                                                   const float* __restrict__ g,
                                                   const float* __restrict__ bta,
                                                   h16* __restrict__ Xh) {
  __shared__ float sm[DM];
  int r = blockIdx.x, c = threadIdx.x;
  size_t idx = (size_t)r * DM + c;
  float v = X[idx] + resid[idx];
  sm[c] = v;
  __syncthreads();
  for (int s = DM / 2; s > 0; s >>= 1) { if (c < s) sm[c] += sm[c + s]; __syncthreads(); }
  float m = sm[0] * (1.f / DM);
  __syncthreads();
  float dv = v - m;
  sm[c] = dv * dv;
  __syncthreads();
  for (int s = DM / 2; s > 0; s >>= 1) { if (c < s) sm[c] += sm[c + s]; __syncthreads(); }
  float var = sm[0] * (1.f / DM);
  float y = dv * rsqrtf(var + 1e-5f) * g[c] + bta[c];
  X[idx] = y;
  if (Xh) Xh[idx] = (h16)y;
}

// ---------------------------------------------------------------------------
// Message-bus roll: roll(+-1, axis=batch[2]) = swap, roll(+-2) = identity.
// ---------------------------------------------------------------------------
__global__ void k_bus_swap(float* __restrict__ X) {
  int i = blockIdx.x * blockDim.x + threadIdx.x;
  if (i >= 16 * DM) return;
  int t = i >> 8, c = i & 255;
  size_t r0 = ((size_t)(HWALL + NCLIP + t)) * DM + c;
  size_t r1 = ((size_t)(LTOT + HWALL + NCLIP + t)) * DM + c;
  float a = X[r0], b = X[r1];
  X[r0] = b;
  X[r1] = a;
}

// ---------------------------------------------------------------------------
// Final: memory = X[:, :HW] plus spatial_shapes + level_start (int bits)
// ---------------------------------------------------------------------------
__global__ void k_final(const float* __restrict__ X, float* __restrict__ out) {
  size_t i = (size_t)blockIdx.x * blockDim.x + threadIdx.x;
  const size_t nmem = (size_t)BS * HWALL * DM;
  if (i < nmem) {
    size_t c = i & 255;
    size_t rt = i >> 8;
    size_t b = rt / HWALL, t = rt - b * HWALL;
    out[i] = X[((size_t)b * LTOT + t) * DM + c];
  } else if (i < nmem + 12) {
    const int vals[12] = {96, 96, 48, 48, 24, 24, 12, 12, 0, 9216, 11520, 12096};
    ((int*)out)[i] = vals[(int)(i - nmem)];
  }
}

// ---------------------------------------------------------------------------
extern "C" void kernel_launch(void* const* d_in, const int* in_sizes, int n_in,
                              void* d_out, int out_size, void* d_ws, size_t ws_size,
                              hipStream_t stream) {
  (void)in_sizes; (void)n_in; (void)out_size; (void)ws_size;
  const float* srcI[4] = {(const float*)d_in[0], (const float*)d_in[2],
                          (const float*)d_in[4], (const float*)d_in[6]};
  const float* posI[4] = {(const float*)d_in[1], (const float*)d_in[3],
                          (const float*)d_in[5], (const float*)d_in[7]};
  const float* level_embed = (const float*)d_in[8];
  const float* memory_bus  = (const float*)d_in[9];
  const float* memory_pos  = (const float*)d_in[10];
  const float* obj_text    = (const float*)d_in[11];
  const float* clip_W      = (const float*)d_in[12];
  const float* clip_b      = (const float*)d_in[13];
  const float* off_W  = (const float*)d_in[14];
  const float* off_b  = (const float*)d_in[15];
  const float* attn_W = (const float*)d_in[16];
  const float* attn_b = (const float*)d_in[17];
  const float* val_W  = (const float*)d_in[18];
  const float* val_b  = (const float*)d_in[19];
  const float* out_W  = (const float*)d_in[20];
  const float* out_b  = (const float*)d_in[21];
  const float* ln1_g  = (const float*)d_in[22];
  const float* ln1_b  = (const float*)d_in[23];
  const float* lin1_W = (const float*)d_in[24];
  const float* lin1_b = (const float*)d_in[25];
  const float* lin2_W = (const float*)d_in[26];
  const float* lin2_b = (const float*)d_in[27];
  const float* ln2_g  = (const float*)d_in[28];
  const float* ln2_b  = (const float*)d_in[29];

  char* w = (char*)d_ws;
  size_t o = 0;
  auto carve = [&](size_t bytes) -> char* {
    char* p = w + o;
    o = (o + bytes + 255) & ~(size_t)255;
    return p;
  };
  float* X    = (float*)carve((size_t)RPAD * DM * 4);
  float* POSB = (float*)carve((size_t)RPAD * DM * 4);
  float* VAL  = (float*)carve((size_t)RPAD * DM * 4);
  float* OFFB = (float*)carve((size_t)RPAD * DM * 4);
  float* TMP  = (float*)carve((size_t)RPAD * DM * 4);
  float* ATT  = (float*)carve((size_t)RPAD * 128 * 4);
  h16* Xh     = (h16*)carve((size_t)RPAD * DM * 2);
  h16* Qh     = (h16*)carve((size_t)RPAD * DM * 2);  // reused as f16 attn-out
  h16* HID    = (h16*)carve((size_t)RPAD * DFF * 2);
  h16* WTval  = (h16*)carve((size_t)NLAYERS * DM * DM * 2);
  h16* WToff  = (h16*)carve((size_t)NLAYERS * DM * DM * 2);
  h16* WTattn = (h16*)carve((size_t)NLAYERS * DM * 128 * 2);
  h16* WTout  = (h16*)carve((size_t)NLAYERS * DM * DM * 2);
  h16* WTlin1 = (h16*)carve((size_t)NLAYERS * DM * DFF * 2);
  h16* WTlin2 = (h16*)carve((size_t)NLAYERS * DFF * DM * 2);

  dim3 b256(256);
  auto g1 = [](int n) { return dim3((unsigned)((n + 255) / 256)); };

  // Weight conversion (f32 -> f16 transposed) for all 6 layers
  for (int li = 0; li < NLAYERS; li++) {
    k_wcvt<<<g1(DM * DM), b256, 0, stream>>>(val_W + (size_t)li * DM * DM,
                                             WTval + (size_t)li * DM * DM, DM, DM);
    k_wcvt<<<g1(DM * DM), b256, 0, stream>>>(off_W + (size_t)li * DM * DM,
                                             WToff + (size_t)li * DM * DM, DM, DM);
    k_wcvt<<<g1(DM * 128), b256, 0, stream>>>(attn_W + (size_t)li * DM * 128,
                                              WTattn + (size_t)li * DM * 128, DM, 128);
    k_wcvt<<<g1(DM * DM), b256, 0, stream>>>(out_W + (size_t)li * DM * DM,
                                             WTout + (size_t)li * DM * DM, DM, DM);
    k_wcvt<<<g1(DM * DFF), b256, 0, stream>>>(lin1_W + (size_t)li * DM * DFF,
                                              WTlin1 + (size_t)li * DM * DFF, DM, DFF);
    k_wcvt<<<g1(DFF * DM), b256, 0, stream>>>(lin2_W + (size_t)li * DFF * DM,
                                              WTlin2 + (size_t)li * DFF * DM, DFF, DM);
  }

  const int nElem = RTOT * DM;
  k_build_pos<<<g1(nElem), b256, 0, stream>>>(posI[0], posI[1], posI[2], posI[3],
                                              level_embed, memory_pos, POSB);
  k_build_x0<<<g1(nElem), b256, 0, stream>>>(srcI[0], srcI[1], srcI[2], srcI[3],
                                             memory_bus, obj_text, clip_W, clip_b, X);

  // GEMM launch geometry: 32 rows x 64 cols per wave
  const dim3 gemmB128(128);   // 4 waves -> 256 cols per block
  const dim3 gemmB64(64);     // 2 waves -> 128 cols per block
  const dim3 gRows(RPAD / 32, 1);
  const dim3 gRows4(RPAD / 32, 4);

  for (int li = 0; li < NLAYERS; li++) {
    k_prep<<<g1(nElem), b256, 0, stream>>>(X, POSB, Xh, Qh);
    // value = X @ val_W + b ; off = Q @ off_W + b ; attn logits = Q @ attn_W + b
    k_gemm<DM, DM, 0><<<gRows, gemmB128, 0, stream>>>(
        Xh, WTval + (size_t)li * DM * DM, val_b + (size_t)li * DM, VAL, nullptr, RTOT);
    k_gemm<DM, DM, 0><<<gRows, gemmB128, 0, stream>>>(
        Qh, WToff + (size_t)li * DM * DM, off_b + (size_t)li * DM, OFFB, nullptr, RTOT);
    k_gemm<DM, 128, 0><<<gRows, gemmB64, 0, stream>>>(
        Qh, WTattn + (size_t)li * DM * 128, attn_b + (size_t)li * 128, ATT, nullptr, RTOT);
    k_softmax<<<g1(RTOT * NH), b256, 0, stream>>>(ATT);
    k_sample<<<dim3(RTOT), b256, 0, stream>>>(VAL, OFFB, ATT, Qh);  // Qh reused as f16 out
    // out projection -> TMP ; LN1 ; FFN ; LN2 ; bus swap
    k_gemm<DM, DM, 0><<<gRows, gemmB128, 0, stream>>>(
        Qh, WTout + (size_t)li * DM * DM, out_b + (size_t)li * DM, TMP, nullptr, RTOT);
    k_layernorm<<<dim3(RTOT), b256, 0, stream>>>(X, TMP, ln1_g + (size_t)li * DM,
                                                 ln1_b + (size_t)li * DM, Xh);
    k_gemm<DM, DFF, 1><<<gRows4, gemmB128, 0, stream>>>(
        Xh, WTlin1 + (size_t)li * DM * DFF, lin1_b + (size_t)li * DFF, nullptr, HID, RTOT);
    k_gemm<DFF, DM, 0><<<gRows, gemmB128, 0, stream>>>(
        HID, WTlin2 + (size_t)li * DFF * DM, lin2_b + (size_t)li * DM, TMP, nullptr, RTOT);
    k_layernorm<<<dim3(RTOT), b256, 0, stream>>>(X, TMP, ln2_g + (size_t)li * DM,
                                                 ln2_b + (size_t)li * DM, nullptr);
    k_bus_swap<<<g1(16 * DM), b256, 0, stream>>>(X);
  }

  size_t ntot = (size_t)BS * HWALL * DM + 12;
  k_final<<<dim3((unsigned)((ntot + 255) / 256)), b256, 0, stream>>>(X, (float*)d_out);
}
